// SimpleBEV_2336462209624
// MI455X (gfx1250) — matrix operand order repaired and checked by reading, compiled, not run on hardware
//
#include <hip/hip_runtime.h>

// ---------- types ----------
typedef __attribute__((ext_vector_type(8)))  float  v8f;
typedef __attribute__((ext_vector_type(16))) __bf16 v16bf;

#define C_FEAT 128
#define HF 32
#define WF 88
#define XDIM 128
#define YDIM 8
#define ZDIM 128
#define S_CAM 6
#define CIN 1024            // C_FEAT * YDIM
#define PADW 130            // 128 + 2 halo
#define NSPAT 16384         // ZDIM * XDIM
#define KCHUNKS 288         // 9 * 1024 / 32

static __device__ inline unsigned short f2bf_bits(float f) {
    union { float f; unsigned u; } v; v.f = f;
    unsigned r = (v.u + 0x7FFFu + ((v.u >> 16) & 1u)) >> 16;   // RNE
    return (unsigned short)r;
}

// ---------- 1) per-camera projection matrices ----------
__global__ void build_mats(const float* __restrict__ K4,   // [6][4][4]
                           const float* __restrict__ c2l,  // [6][4][4]
                           float* __restrict__ mats) {
    int s = threadIdx.x;
    if (s >= S_CAM) return;
    const float sx = (float)WF / 704.0f, sy = (float)HF / 256.0f;
    const float* K = K4 + s * 16;
    const float* M = c2l + s * 16;
    float fx = K[0] * sx, cx = K[2] * sx;
    float fy = K[5] * sy, cy = K[6] * sy;
    float R[3][3], t[3];
    for (int i = 0; i < 3; ++i) { for (int j = 0; j < 3; ++j) R[i][j] = M[i*4+j]; t[i] = M[i*4+3]; }
    float inv[3][4];
    for (int i = 0; i < 3; ++i) {
        inv[i][0] = R[0][i]; inv[i][1] = R[1][i]; inv[i][2] = R[2][i];
        inv[i][3] = -(R[0][i]*t[0] + R[1][i]*t[1] + R[2][i]*t[2]);
    }
    float* o = mats + s * 12;
    for (int j = 0; j < 4; ++j) o[j]     = fx * inv[0][j] + cx * inv[2][j];
    for (int j = 0; j < 4; ++j) o[4 + j] = fy * inv[1][j] + cy * inv[2][j];
    for (int j = 0; j < 4; ++j) o[8 + j] = inv[2][j];
}

// ---------- 2) zero padded BEV buffer ----------
__global__ void zero_fp(uint4* __restrict__ p) {
    uint4 z; z.x = 0u; z.y = 0u; z.z = 0u; z.w = 0u;
    p[blockIdx.x * blockDim.x + threadIdx.x] = z;
}

// ---------- 3) weight swizzle: [cout][cin][3][3] f32 ->
//             per-K-chunk WMMA-A fragment order bf16: [chunk=288][mt=8][lane=32][slot=16]
__global__ void reorder_w(const float* __restrict__ w, unsigned short* __restrict__ Wr2) {
    int idx = blockIdx.x * blockDim.x + threadIdx.x;
    if (idx >= 9 * CIN * C_FEAT) return;
    int chunk = idx >> 12;            // / 4096
    int rem12 = idx & 4095;
    int mt    = rem12 >> 9;           // cout tile
    int lane  = (rem12 >> 4) & 31;
    int slot  = rem12 & 15;
    int r     = chunk >> 5;           // 3x3 tap
    int cidx  = chunk & 31;           // cin block of 32
    int hh    = lane >> 4, j = lane & 15;
    int p     = slot >> 1, e = slot & 1;
    // contiguous k order matching B staging: k = hh*16 + slot
    int k_local = (hh << 4) | slot;
    int cin   = (cidx << 5) + k_local;
    int cout  = (mt << 4) + j;
    Wr2[idx] = f2bf_bits(w[(cout * CIN + cin) * 9 + r]);
}

// ---------- 4) project voxels, bilinear sample, average cams, write padded bf16 BEV ----------
__global__ __launch_bounds__(256) void project_sample(const float* __restrict__ cam_feat,
                                                      const float* __restrict__ mats,
                                                      unsigned short* __restrict__ fp) {
    int n  = blockIdx.x * blockDim.x + threadIdx.x;   // voxel id, 131072 total
    int xi = n & (XDIM - 1);
    int yi = (n >> 7) & (YDIM - 1);
    int zi = n >> 10;
    const float vx = 100.0f / XDIM, vy = 10.0f / YDIM, vz = 100.0f / ZDIM;
    float x = xi * vx - 50.0f + 0.5f * vx;
    float y = yi * vy - 4.0f  + 0.5f * vy;
    float z = zi * vz - 50.0f + 0.5f * vz;

    int   offs[S_CAM][4];
    float wts [S_CAM][4];
    bool  vcam[S_CAM];
#pragma unroll
    for (int cam = 0; cam < S_CAM; ++cam) {
        const float* m = mats + cam * 12;
        float px = m[0]*x + m[1]*y + m[2]*z  + m[3];
        float py = m[4]*x + m[5]*y + m[6]*z  + m[7];
        float zc = m[8]*x + m[9]*y + m[10]*z + m[11];
        float denom = fmaxf(zc, 1e-6f);
        float xs = px / denom, ys = py / denom;
        bool valid = (xs > -0.5f) && (xs < WF - 0.5f) && (ys > -0.5f) && (ys < HF - 0.5f) && (zc > 0.0f);
        float gx = fminf(fmaxf(2.0f * xs / (WF - 1.0f) - 1.0f, -2.0f), 2.0f);
        float gy = fminf(fmaxf(2.0f * ys / (HF - 1.0f) - 1.0f, -2.0f), 2.0f);
        float ixf = ((gx + 1.0f) * WF - 1.0f) * 0.5f;
        float iyf = ((gy + 1.0f) * HF - 1.0f) * 0.5f;
        float x0f = floorf(ixf), y0f = floorf(iyf);
        float wx1 = ixf - x0f,  wy1 = iyf - y0f;
#pragma unroll
        for (int ty = 0; ty < 2; ++ty)
#pragma unroll
            for (int tx = 0; tx < 2; ++tx) {
                float xf = x0f + tx, yf = y0f + ty;
                bool inb = (xf >= 0.0f) && (xf <= WF - 1.0f) && (yf >= 0.0f) && (yf <= HF - 1.0f);
                int xc = (int)fminf(fmaxf(xf, 0.0f), WF - 1.0f);
                int yc = (int)fminf(fmaxf(yf, 0.0f), HF - 1.0f);
                float wgt = (tx ? wx1 : 1.0f - wx1) * (ty ? wy1 : 1.0f - wy1);
                offs[cam][ty*2+tx] = yc * WF + xc;
                wts [cam][ty*2+tx] = (inb && valid) ? wgt : 0.0f;
            }
        vcam[cam] = valid;
    }

#pragma unroll 4
    for (int c = 0; c < C_FEAT; ++c) {
        float s = 0.0f, cnt = 0.0f;
#pragma unroll
        for (int cam = 0; cam < S_CAM; ++cam) {
            if (vcam[cam]) {
                const float* f = cam_feat + (size_t)(cam * C_FEAT + c) * (HF * WF);
                float v = f[offs[cam][0]] * wts[cam][0] + f[offs[cam][1]] * wts[cam][1]
                        + f[offs[cam][2]] * wts[cam][2] + f[offs[cam][3]] * wts[cam][3];
                if (fabsf(v) > 0.0f) { s += v; cnt += 1.0f; }
            }
        }
        float outv = s / (1e-6f + cnt);
        // padded layout [cin=c*8+y][1+z][1+x]
        fp[((size_t)(c * YDIM + yi) * PADW + (zi + 1)) * PADW + (xi + 1)] = f2bf_bits(outv);
    }
}

// ---------- 5) implicit-GEMM 3x3 conv via WMMA bf16 ----------
// Block tile: 128 cout x 64 spatial; 8 waves, each wave = 16 cout x 64 (4 accumulators).
// A fragments stream pre-swizzled from global (L2-resident); B double-buffered in LDS,
// one barrier per K-step, 4 WMMAs per wave per barrier.
__global__ __launch_bounds__(256) void conv_gemm(const unsigned short* __restrict__ Wr2,
                                                 const unsigned short* __restrict__ fp,
                                                 float* __restrict__ out) {
    __shared__ __align__(32) unsigned short Bs[2][4][32][16];   // [buf][nt][lane][slot] = 8 KB
    int t    = threadIdx.x;
    int lane = t & 31;
    int wv   = t >> 5;                    // wave id = cout tile (0..7)
    int h    = blockIdx.x >> 1;           // z row
    int w0   = (blockIdx.x & 1) << 6;     // x tile base (64 wide)

    // B staging map: thread owns k-pair (2 cins) x 4 cols; packs 2 bf16/dword
    int k0 = (t >> 4) << 1;               // even k (0..30)
    int n0 = (t & 15) << 2;               // 0..60 step 4
    int laneB = (k0 >> 4) << 4;           // lanes 0-15 hold K 0-15, lanes 16-31 hold K 16-31
    int s0 = k0 & 15;                     // even slot -> dword-aligned pair

    v8f acc0 = {}, acc1 = {}, acc2 = {}, acc3 = {};

    auto params = [&](int kk, int& c0, int& dh, int& dw) {
        int r = kk >> 5;                  // 3x3 tap (0..8)
        dh = (r >= 6) ? 2 : ((r >= 3) ? 1 : 0);
        dw = r - dh * 3;
        c0 = (kk & 31) << 5;              // cin base
    };

    auto stageB = [&](int c0, int dh, int dw, int buf) {
        const unsigned short* b0 = fp + ((size_t)(c0 + k0)     * PADW + (h + dh)) * PADW + (w0 + dw) + n0;
        const unsigned short* b1 = fp + ((size_t)(c0 + k0 + 1) * PADW + (h + dh)) * PADW + (w0 + dw) + n0;
#pragma unroll
        for (int dn = 0; dn < 4; ++dn) {
            int n = n0 + dn;
            unsigned v = (unsigned)b0[dn] | ((unsigned)b1[dn] << 16);
            *reinterpret_cast<unsigned*>(&Bs[buf][n >> 4][laneB + (n & 15)][s0]) = v;
        }
    };

    { int c0, dh, dw; params(0, c0, dh, dw); stageB(c0, dh, dw, 0); }
    __syncthreads();

    for (int kk = 0; kk < KCHUNKS; ++kk) {
        if (kk + 1 < KCHUNKS) {
            int c0n, dhn, dwn; params(kk + 1, c0n, dhn, dwn);
            stageB(c0n, dhn, dwn, (kk + 1) & 1);
        }

        // A fragment: 32 contiguous bytes per lane, pre-swizzled in global
        const unsigned short* srcA = Wr2 + (((size_t)kk) << 12) + (wv << 9) + (lane << 4);
        v16bf a = *reinterpret_cast<const v16bf*>(srcA);
        if (kk + 4 < KCHUNKS)
            __builtin_prefetch(Wr2 + (((size_t)(kk + 4)) << 12) + (wv << 9) + (lane << 4), 0, 0);

        int buf = kk & 1;
        v16bf b0f = *reinterpret_cast<const v16bf*>(&Bs[buf][0][lane][0]);
        v16bf b1f = *reinterpret_cast<const v16bf*>(&Bs[buf][1][lane][0]);
        v16bf b2f = *reinterpret_cast<const v16bf*>(&Bs[buf][2][lane][0]);
        v16bf b3f = *reinterpret_cast<const v16bf*>(&Bs[buf][3][lane][0]);
        acc0 = __builtin_amdgcn_wmma_f32_16x16x32_bf16(false, a, false, b0f, (short)0, acc0, false, false);
        acc1 = __builtin_amdgcn_wmma_f32_16x16x32_bf16(false, a, false, b1f, (short)0, acc1, false, false);
        acc2 = __builtin_amdgcn_wmma_f32_16x16x32_bf16(false, a, false, b2f, (short)0, acc2, false, false);
        acc3 = __builtin_amdgcn_wmma_f32_16x16x32_bf16(false, a, false, b3f, (short)0, acc3, false, false);
        __syncthreads();
    }

    // C/D 16x16 f32 layout: lanes 0-15 -> M=r, N=lane; lanes 16-31 -> M=r+8
    int coutBase = wv * 16 + ((lane >> 4) << 3);
    int n0g = w0 + (lane & 15);
#pragma unroll
    for (int ri = 0; ri < 8; ++ri) {
        size_t base = (size_t)(coutBase + ri) * NSPAT + h * XDIM + n0g;
        out[base]      = acc0[ri];
        out[base + 16] = acc1[ri];
        out[base + 32] = acc2[ri];
        out[base + 48] = acc3[ri];
    }
}

// ---------- 6) per-channel instance norm + exact GELU, in place on d_out ----------
__global__ __launch_bounds__(256) void norm_gelu(float* __restrict__ out) {
    __shared__ float ssum[256], ssq[256];
    int c = blockIdx.x;
    float* p = out + (size_t)c * NSPAT;
    float s = 0.0f, q = 0.0f;
    for (int i = threadIdx.x; i < NSPAT; i += 256) { float v = p[i]; s += v; q += v * v; }
    ssum[threadIdx.x] = s; ssq[threadIdx.x] = q;
    __syncthreads();
    for (int st = 128; st > 0; st >>= 1) {
        if (threadIdx.x < st) { ssum[threadIdx.x] += ssum[threadIdx.x + st]; ssq[threadIdx.x] += ssq[threadIdx.x + st]; }
        __syncthreads();
    }
    float mu  = ssum[0] * (1.0f / NSPAT);
    float var = ssq[0]  * (1.0f / NSPAT) - mu * mu;
    float rs  = rsqrtf(var + 1e-5f);
    for (int i = threadIdx.x; i < NSPAT; i += 256) {
        float v = (p[i] - mu) * rs;
        p[i] = 0.5f * v * (1.0f + erff(v * 0.70710678118f));
    }
}

// ---------- launch ----------
extern "C" void kernel_launch(void* const* d_in, const int* in_sizes, int n_in,
                              void* d_out, int out_size, void* d_ws, size_t ws_size,
                              hipStream_t stream) {
    const float* cam_feat = (const float*)d_in[0];  // [6,128,32,88]
    const float* Kmat     = (const float*)d_in[4];  // cam_intrinsic [1,6,4,4]
    const float* c2l      = (const float*)d_in[5];  // camera2lidar  [1,6,4,4]
    const float* convw    = (const float*)d_in[6];  // [128,1024,3,3]
    float* out = (float*)d_out;                     // [1,128,128,128] f32

    char* ws = (char*)d_ws;
    float*          mats = (float*)ws;                                   // 288 B
    unsigned short* Wr2  = (unsigned short*)(ws + 512);                  // 288*4096*2 = 2,359,296 B
    unsigned short* fpad = (unsigned short*)(ws + 512 + 2359296);        // 1024*130*130*2 = 34,611,200 B

    build_mats<<<1, 32, 0, stream>>>(Kmat, c2l, mats);
    zero_fp<<<(1024 * PADW * PADW * 2 / 16) / 256, 256, 0, stream>>>((uint4*)fpad);
    reorder_w<<<(9 * CIN * C_FEAT + 255) / 256, 256, 0, stream>>>(convw, Wr2);
    project_sample<<<(ZDIM * YDIM * XDIM) / 256, 256, 0, stream>>>(cam_feat, mats, fpad);
    conv_gemm<<<ZDIM * (XDIM / 64), 256, 0, stream>>>(Wr2, fpad, out);
    norm_gelu<<<C_FEAT, 256, 0, stream>>>(out);
}
